// SAMCGA_61821759259390
// MI455X (gfx1250) — matrix operand order repaired and checked by reading, not compiled, not run on hardware
//
#include <hip/hip_runtime.h>
#include <hip/hip_bf16.h>

#define DIM 2048
#define NHEAD 8
#define HDIM 256
#define BATCH 256
#define KC 128           // K-chunk staged per LDS buffer
#define APAD 8           // row padding (bf16 elems) -> conflict-free ds_load_b128

typedef __attribute__((ext_vector_type(16))) __bf16 v16bf;
typedef __attribute__((ext_vector_type(8)))  float  v8f;

union BFrag { v16bf v; uint4 q[2]; };

__device__ __forceinline__ unsigned short f2bf(float f) {
    unsigned int u = __float_as_uint(f);
    u += 0x7FFFu + ((u >> 16) & 1u);   // round-to-nearest-even
    return (unsigned short)(u >> 16);
}

__device__ __forceinline__ float gelu_t(float x) {
    const float c0 = 0.7978845608028654f; // sqrt(2/pi)
    return 0.5f * x * (1.f + tanhf(c0 * (x + 0.044715f * x * x * x)));
}

// CDNA5 async global->LDS copy (ASYNCcnt-tracked), 16 bytes per lane.
__device__ __forceinline__ void async_copy16(unsigned int lds_addr,
                                             const unsigned short* g) {
    asm volatile("global_load_async_to_lds_b128 %0, %1, off"
                 :: "v"(lds_addr), "v"(g) : "memory");
}
#define WAIT_ASYNC(n) asm volatile("s_wait_asynccnt %0" :: "n"(n) : "memory")

// ---------------------------------------------------------------------------
// 1) mem [B, DIM, 4,4,4] -> pooled bf16 [B, DIM]  (mean over 64 contiguous)
// ---------------------------------------------------------------------------
__global__ __launch_bounds__(256)
void pool_mem(const float* __restrict__ mem, unsigned short* __restrict__ pooled) {
    int i = blockIdx.x * blockDim.x + threadIdx.x;   // one (b,d) per thread
    const float4* p = (const float4*)(mem + (size_t)i * 64);
    float s = 0.f;
#pragma unroll
    for (int j = 0; j < 16; ++j) { float4 t = p[j]; s += t.x + t.y + t.z + t.w; }
    pooled[i] = f2bf(s * (1.f / 64.f));
}

// ---------------------------------------------------------------------------
// 2) elementwise f32 -> bf16 (8 elements / thread, vectorized)
// ---------------------------------------------------------------------------
__global__ __launch_bounds__(256)
void cvt_f32_bf16(const float* __restrict__ src, unsigned short* __restrict__ dst,
                  int n8) {
    int i = blockIdx.x * blockDim.x + threadIdx.x;
    if (i >= n8) return;
    const float4* s = (const float4*)(src + (size_t)i * 8);
    float4 a = s[0], b = s[1];
    union { uint4 q; unsigned short u[8]; } o;
    o.u[0] = f2bf(a.x); o.u[1] = f2bf(a.y); o.u[2] = f2bf(a.z); o.u[3] = f2bf(a.w);
    o.u[4] = f2bf(b.x); o.u[5] = f2bf(b.y); o.u[6] = f2bf(b.z); o.u[7] = f2bf(b.w);
    *(uint4*)(dst + (size_t)i * 8) = o.q;
}

// ---------------------------------------------------------------------------
// 3) k2 branch: red = f_age . rW + rb ; k2 = f_age + eW0*red + eW1*gap + eb
// ---------------------------------------------------------------------------
__global__ __launch_bounds__(256)
void k2_kernel(const float* __restrict__ f_age, const float* __restrict__ age_gap,
               const float* __restrict__ rW, const float* __restrict__ rb,
               const float* __restrict__ eW, const float* __restrict__ eb,
               float* __restrict__ k2) {
    int b = blockIdx.x;
    __shared__ float sred[8];
    const float* x = f_age + (size_t)b * DIM;
    float s = 0.f;
    for (int d = threadIdx.x; d < DIM; d += 256) s += x[d] * rW[d];
#pragma unroll
    for (int o = 16; o; o >>= 1) s += __shfl_xor(s, o, 32);
    if ((threadIdx.x & 31) == 0) sred[threadIdx.x >> 5] = s;
    __syncthreads();
    if (threadIdx.x == 0) {
        float t = 0.f;
#pragma unroll
        for (int i = 0; i < 8; ++i) t += sred[i];
        sred[0] = t + rb[0];
    }
    __syncthreads();
    float red = sred[0];
    float g = age_gap[b];
    for (int d = threadIdx.x; d < DIM; d += 256)
        k2[(size_t)b * DIM + d] = x[d] + eW[2 * d] * red + eW[2 * d + 1] * g + eb[d];
}

// ---------------------------------------------------------------------------
// 4) GEMM: C[M,N] = A[M,K] @ W[N,K]^T, bf16 in / f32 accumulate.
//    Block = 4 waves -> 64x64 output tile.  A tile (64xKC) staged once per
//    block in double-buffered LDS via global_load_async_to_lds_b128 and read
//    as fragments with ds_load_b128 (row pitch KC+8 -> conflict-free).
//    B strip double-buffered in registers.  16 WMMAs per chunk.
// ---------------------------------------------------------------------------
__global__ __launch_bounds__(128)
void gemm_rowT_bf16(const unsigned short* __restrict__ A,
                    const unsigned short* __restrict__ W,
                    float* __restrict__ C, int N, int K) {
    __shared__ unsigned short sA[2][64][KC + APAD];   // 2 x 17 KB

    const int t    = threadIdx.x;          // 0..127
    const int lane = t & 31;
    const int wave = t >> 5;
    const int r = lane & 15;               // row-in-tile (A) / col-in-tile (B)
    const int h = lane >> 4;               // half selector
    const int n0 = blockIdx.x * 64 + wave * 16;
    const int m0 = blockIdx.y * 64;
    const int NC = K / KC;

    // async staging: thread t copies half-row (64 bf16 = 128 B) of the A tile
    const int srow = t >> 1, shalf = t & 1;
    const unsigned short* gbase = A + (size_t)(m0 + srow) * K + shalf * 64;
    unsigned int lbase =
        (unsigned int)(uintptr_t)(&sA[0][srow][shalf * 64]);
    const unsigned int lpitch = (unsigned int)(sizeof(unsigned short) * 64 * (KC + APAD));

    v8f acc[4];
#pragma unroll
    for (int mt = 0; mt < 4; ++mt) acc[mt] = (v8f){0.f,0.f,0.f,0.f,0.f,0.f,0.f,0.f};

    const unsigned short* wrow = W + (size_t)(n0 + r) * K + 8 * h;

    // preload chunk 0 (LDS A) and its B strip (registers)
#pragma unroll
    for (int i = 0; i < 8; ++i) async_copy16(lbase + i * 16, gbase + i * 8);
    BFrag Bc[4], Bn[4];
#pragma unroll
    for (int i = 0; i < 4; ++i) {
        Bc[i].q[0] = *(const uint4*)(wrow + 32 * i);
        Bc[i].q[1] = *(const uint4*)(wrow + 32 * i + 16);
    }

    for (int c = 0; c < NC; ++c) {
        const int bb = c & 1;
        if (c + 1 < NC) {
            // stage next A chunk into the other buffer + prefetch next B strip
            const unsigned short* gn = gbase + (size_t)(c + 1) * KC;
            unsigned int ln = lbase + (bb ? 0u : lpitch);
#pragma unroll
            for (int i = 0; i < 8; ++i) async_copy16(ln + i * 16, gn + i * 8);
#pragma unroll
            for (int i = 0; i < 4; ++i) {
                Bn[i].q[0] = *(const uint4*)(wrow + (c + 1) * KC + 32 * i);
                Bn[i].q[1] = *(const uint4*)(wrow + (c + 1) * KC + 32 * i + 16);
            }
            WAIT_ASYNC(8);     // chunk c landed; chunk c+1 still in flight
        } else {
            WAIT_ASYNC(0);
        }
        __syncthreads();       // chunk c visible block-wide

#pragma unroll
        for (int ksi = 0; ksi < 4; ++ksi) {
#pragma unroll
            for (int mt = 0; mt < 4; ++mt) {
                const unsigned short* ap = &sA[bb][mt * 16 + r][ksi * 32 + 8 * h];
                BFrag af;
                af.q[0] = *(const uint4*)(ap);
                af.q[1] = *(const uint4*)(ap + 16);
                acc[mt] = __builtin_amdgcn_wmma_f32_16x16x32_bf16(
                    false, af.v, false, Bc[ksi].v, (short)0, acc[mt], false, false);
            }
        }
        __syncthreads();       // all waves done reading buf bb before it is re-staged
#pragma unroll
        for (int i = 0; i < 4; ++i) Bc[i] = Bn[i];
    }

    // C/D layout: VGPR rr, lanes 0-15 -> M = rr, lanes 16-31 -> M = 8+rr
    const int col = n0 + r;
#pragma unroll
    for (int mt = 0; mt < 4; ++mt) {
#pragma unroll
        for (int rr = 0; rr < 8; ++rr) {
            int row = m0 + mt * 16 + 8 * h + rr;
            C[(size_t)row * N + col] = acc[mt][rr];
        }
    }
}

// ---------------------------------------------------------------------------
// 5) fused: per-head LN -> dual 8x8 attention -> conv1d(3) -> LNs -> gelu ->
//    final LN(2048) -> gelu.  One block per batch row; wave w owns head w.
// ---------------------------------------------------------------------------
__device__ __forceinline__ void wave_ln_load(const float* __restrict__ src,
                                             float* __restrict__ dst,
                                             const float* __restrict__ lnw,
                                             const float* __restrict__ lnb, int lane) {
    float x[8]; float s = 0.f, s2 = 0.f;
#pragma unroll
    for (int j = 0; j < 8; ++j) { x[j] = src[lane + 32 * j]; s += x[j]; s2 += x[j] * x[j]; }
#pragma unroll
    for (int o = 16; o; o >>= 1) { s += __shfl_xor(s, o, 32); s2 += __shfl_xor(s2, o, 32); }
    float mu  = s * (1.f / 256.f);
    float inv = rsqrtf(s2 * (1.f / 256.f) - mu * mu + 1e-5f);
#pragma unroll
    for (int j = 0; j < 8; ++j) { int e = lane + 32 * j; dst[e] = (x[j] - mu) * inv * lnw[e] + lnb[e]; }
}

__device__ __forceinline__ void wave_ln_reg(float (&x)[8],
                                            const float* __restrict__ lnw,
                                            const float* __restrict__ lnb, int lane) {
    float s = 0.f, s2 = 0.f;
#pragma unroll
    for (int j = 0; j < 8; ++j) { s += x[j]; s2 += x[j] * x[j]; }
#pragma unroll
    for (int o = 16; o; o >>= 1) { s += __shfl_xor(s, o, 32); s2 += __shfl_xor(s2, o, 32); }
    float mu  = s * (1.f / 256.f);
    float inv = rsqrtf(s2 * (1.f / 256.f) - mu * mu + 1e-5f);
#pragma unroll
    for (int j = 0; j < 8; ++j) { int e = lane + 32 * j; x[j] = (x[j] - mu) * inv * lnw[e] + lnb[e]; }
}

__global__ __launch_bounds__(256)
void fused_attn(const float* __restrict__ qv, const float* __restrict__ k1,
                const float* __restrict__ k2,
                const float* __restrict__ ln_w, const float* __restrict__ ln_b,
                const float* __restrict__ ln2_w, const float* __restrict__ ln2_b,
                const float* __restrict__ c1W, const float* __restrict__ c2W,
                float* __restrict__ out) {
    __shared__ float s_q[DIM], s_k1[DIM], s_k2[DIM], s_v[DIM], s_o1[DIM], s_o2[DIM];
    __shared__ float rbuf[16];

    const int b = blockIdx.x;
    const int lane = threadIdx.x & 31, wave = threadIdx.x >> 5;
    const int hoff = wave * HDIM;
    const float* qrow = qv + (size_t)b * (2 * DIM);

    // per-head layernorms of q, k1, k2; copy v
    wave_ln_load(qrow + hoff,                 s_q  + hoff, ln_w, ln_b, lane);
    wave_ln_load(k1 + (size_t)b * DIM + hoff, s_k1 + hoff, ln_w, ln_b, lane);
    wave_ln_load(k2 + (size_t)b * DIM + hoff, s_k2 + hoff, ln_w, ln_b, lane);
#pragma unroll
    for (int j = 0; j < 8; ++j) {
        int e = lane + 32 * j;
        s_v[hoff + e] = qrow[DIM + hoff + e];
    }
    __syncthreads();

    // dual attention, head l = wave; 8x8 scores, softmax over s
    float qreg[8];
#pragma unroll
    for (int j = 0; j < 8; ++j) qreg[j] = s_q[hoff + lane + 32 * j];
    float sc1[8], sc2[8];
#pragma unroll
    for (int s = 0; s < 8; ++s) {
        float d1 = 0.f, d2 = 0.f;
#pragma unroll
        for (int j = 0; j < 8; ++j) {
            int e = s * HDIM + lane + 32 * j;
            d1 += qreg[j] * s_k1[e];
            d2 += qreg[j] * s_k2[e];
        }
#pragma unroll
        for (int o = 16; o; o >>= 1) { d1 += __shfl_xor(d1, o, 32); d2 += __shfl_xor(d2, o, 32); }
        sc1[s] = d1 * 0.0625f;   // 1/sqrt(256)
        sc2[s] = d2 * 0.0625f;
    }
    float mx1 = sc1[0], mx2 = sc2[0];
#pragma unroll
    for (int s = 1; s < 8; ++s) { mx1 = fmaxf(mx1, sc1[s]); mx2 = fmaxf(mx2, sc2[s]); }
    float t1 = 0.f, t2 = 0.f;
#pragma unroll
    for (int s = 0; s < 8; ++s) {
        sc1[s] = expf(sc1[s] - mx1); t1 += sc1[s];
        sc2[s] = expf(sc2[s] - mx2); t2 += sc2[s];
    }
    float i1 = 1.f / t1, i2 = 1.f / t2;
#pragma unroll
    for (int s = 0; s < 8; ++s) { sc1[s] *= i1; sc2[s] *= i2; }
#pragma unroll
    for (int j = 0; j < 8; ++j) {
        int e = lane + 32 * j;
        float o1 = 0.f, o2 = 0.f;
#pragma unroll
        for (int s = 0; s < 8; ++s) {
            float vv = s_v[s * HDIM + e];
            o1 += sc1[s] * vv; o2 += sc2[s] * vv;
        }
        s_o1[hoff + e] = o1; s_o2[hoff + e] = o2;
    }
    __syncthreads();

    // conv1d (channels=8, k=3, pad=1), wave = output channel
    float w1[24], w2[24];
#pragma unroll
    for (int i = 0; i < 24; ++i) { w1[i] = c1W[wave * 24 + i]; w2[i] = c2W[wave * 24 + i]; }
    float y1[8], y2[8];
#pragma unroll
    for (int j = 0; j < 8; ++j) {
        int p = lane + 32 * j;
        float a1 = 0.f, a2 = 0.f;
#pragma unroll
        for (int c = 0; c < 8; ++c) {
#pragma unroll
            for (int t = 0; t < 3; ++t) {
                int pp = p + t - 1;
                if (pp >= 0 && pp < HDIM) {
                    a1 += s_o1[c * HDIM + pp] * w1[c * 3 + t];
                    a2 += s_o2[c * HDIM + pp] * w2[c * 3 + t];
                }
            }
        }
        y1[j] = a1; y2[j] = a2;
    }
    wave_ln_reg(y1, ln_w, ln_b, lane);
    wave_ln_reg(y2, ln_w, ln_b, lane);

    // gelu(ln(y1 + y2 + v))
    float g[8];
#pragma unroll
    for (int j = 0; j < 8; ++j) g[j] = y1[j] + y2[j] + s_v[hoff + lane + 32 * j];
    wave_ln_reg(g, ln_w, ln_b, lane);
#pragma unroll
    for (int j = 0; j < 8; ++j) g[j] = gelu_t(g[j]);

    // final LN over 2048 + gelu
    float s = 0.f, s2 = 0.f;
#pragma unroll
    for (int j = 0; j < 8; ++j) { s += g[j]; s2 += g[j] * g[j]; }
#pragma unroll
    for (int o = 16; o; o >>= 1) { s += __shfl_xor(s, o, 32); s2 += __shfl_xor(s2, o, 32); }
    if (lane == 0) { rbuf[wave] = s; rbuf[8 + wave] = s2; }
    __syncthreads();
    float ts = 0.f, ts2 = 0.f;
#pragma unroll
    for (int i = 0; i < 8; ++i) { ts += rbuf[i]; ts2 += rbuf[8 + i]; }
    float mu  = ts * (1.f / 2048.f);
    float inv = rsqrtf(ts2 * (1.f / 2048.f) - mu * mu + 1e-5f);
#pragma unroll
    for (int j = 0; j < 8; ++j) {
        int d = hoff + lane + 32 * j;
        out[(size_t)b * DIM + d] = gelu_t((g[j] - mu) * inv * ln2_w[d] + ln2_b[d]);
    }
}

// ---------------------------------------------------------------------------
extern "C" void kernel_launch(void* const* d_in, const int* in_sizes, int n_in,
                              void* d_out, int out_size, void* d_ws, size_t ws_size,
                              hipStream_t stream) {
    const float* f_ad    = (const float*)d_in[0];
    const float* f_age   = (const float*)d_in[1];
    const float* mem     = (const float*)d_in[2];
    const float* age_gap = (const float*)d_in[3];
    const float* qv_W    = (const float*)d_in[4];
    const float* k1_W    = (const float*)d_in[5];
    /* d_in[6] = k2_W: unused by reference */
    const float* ln_w    = (const float*)d_in[7];
    const float* ln_b    = (const float*)d_in[8];
    const float* ln2_w   = (const float*)d_in[9];
    const float* ln2_b   = (const float*)d_in[10];
    const float* rW      = (const float*)d_in[11];
    const float* rb      = (const float*)d_in[12];
    const float* eW      = (const float*)d_in[13];
    const float* eb      = (const float*)d_in[14];
    const float* c1W     = (const float*)d_in[15];
    const float* c2W     = (const float*)d_in[16];
    float* out = (float*)d_out;

    // workspace layout (f32 intermediates, then bf16 copies)
    float* ws     = (float*)d_ws;
    float* k1buf  = ws;                                  // 256*2048 f32
    float* k2buf  = ws + (size_t)BATCH * DIM;            // 256*2048 f32
    float* qvbuf  = ws + (size_t)2 * BATCH * DIM;        // 256*4096 f32
    unsigned short* poolb = (unsigned short*)(ws + (size_t)4 * BATCH * DIM);
    unsigned short* fadb  = poolb + (size_t)BATCH * DIM;          // 256*2048
    unsigned short* k1Wb  = fadb  + (size_t)BATCH * DIM;          // 2048*2048
    unsigned short* qvWb  = k1Wb  + (size_t)DIM * DIM;            // 4096*2048

    // bf16 staging (bandwidth-bound, one pass)
    pool_mem<<<dim3((BATCH * DIM) / 256), 256, 0, stream>>>(mem, poolb);
    cvt_f32_bf16<<<dim3((BATCH * DIM / 8) / 256), 256, 0, stream>>>(f_ad, fadb,
                                                                    BATCH * DIM / 8);
    cvt_f32_bf16<<<dim3((DIM * DIM / 8) / 256), 256, 0, stream>>>(k1_W, k1Wb,
                                                                  DIM * DIM / 8);
    cvt_f32_bf16<<<dim3((2 * DIM * DIM / 8) / 256), 256, 0, stream>>>(qv_W, qvWb,
                                                                      2 * DIM * DIM / 8);
    k2_kernel<<<dim3(BATCH), 256, 0, stream>>>(f_age, age_gap, rW, rb, eW, eb, k2buf);

    // WMMA GEMMs (async-LDS pipelined)
    gemm_rowT_bf16<<<dim3(DIM / 64, BATCH / 64), 128, 0, stream>>>(poolb, k1Wb,
                                                                   k1buf, DIM, DIM);
    gemm_rowT_bf16<<<dim3((2 * DIM) / 64, BATCH / 64), 128, 0, stream>>>(fadb, qvWb,
                                                                         qvbuf, 2 * DIM, DIM);

    // epilogue
    fused_attn<<<dim3(BATCH), 256, 0, stream>>>(qvbuf, k1buf, k2buf, ln_w, ln_b,
                                                ln2_w, ln2_b, c1W, c2W, out);
}
